// ProbSparseAttention_1726576856581
// MI455X (gfx1250) — matrix operand order repaired and checked by reading, not compiled, not run on hardware
//
#include <hip/hip_runtime.h>
#include <hip/hip_bf16.h>

typedef __bf16 bf16_t;
typedef __attribute__((ext_vector_type(16))) __bf16 v16bf;
typedef __attribute__((ext_vector_type(8)))  float  v8f;
typedef __attribute__((ext_vector_type(4)))  unsigned int v4u;
typedef __attribute__((ext_vector_type(8)))  int v8i;
typedef __attribute__((ext_vector_type(4)))  int v4i;

#define BB   2
#define SS   2048
#define DMM  1024
#define HH   16
#define DKK  64
#define HDK  (HH*DKK)      // 1024
#define UU   40
#define SKK  40
#define BHT  (BB*HH)       // 32
#define RR   (BB*SS)       // 4096

// GEMM block tiling
#define BM 64
#define BN 256
#define BK 32

// ---------------------------------------------------------------------------
// TDM: issue a 2D tensor_load_to_lds. Loads a tile of tile1 rows x tile0
// elements (bf16) from a row-major tensor (row stride = strideElems) into
// contiguous LDS at lds_addr. Rows/cols beyond (dim1, dim0) are zero-filled.
// D# bit packing per CDNA5 ISA 8.3/8.4 (group0: count/lds/global/type,
// group1: data_size, tensor dims, tile dims, dim0 stride).
// ---------------------------------------------------------------------------
__device__ __forceinline__ void tdm_load_2d(unsigned lds_addr, const bf16_t* gptr,
                                            long long dim0, long long dim1,
                                            long long strideElems, int tile0, int tile1)
{
    unsigned long long ga = (unsigned long long)(uintptr_t)gptr;
    v4u g0;
    g0.x = 1u;                                                  // count = 1 valid descriptor
    g0.y = lds_addr;                                            // LDS byte address
    g0.z = (unsigned)(ga & 0xFFFFFFFFu);                        // global_addr[31:0]
    g0.w = (unsigned)((ga >> 32) & 0x1FFFFFFu) | (2u << 30);    // global_addr[56:32] | type=2

    unsigned d0 = (unsigned)dim0, d1 = (unsigned)dim1;
    unsigned long long st = (unsigned long long)strideElems;
    v8i g1;
    g1[0] = (int)(1u << 16);                                    // data_size = 1 (2 bytes)
    g1[1] = (int)((d0 & 0xFFFFu) << 16);                        // tensor_dim0[15:0]
    g1[2] = (int)(((d0 >> 16) & 0xFFFFu) | ((d1 & 0xFFFFu) << 16));      // dim0 hi | dim1 lo
    g1[3] = (int)(((d1 >> 16) & 0xFFFFu) | (((unsigned)tile0 & 0xFFFFu) << 16)); // dim1 hi | tile_dim0
    g1[4] = (int)((unsigned)tile1 & 0xFFFFu);                   // tile_dim1 (tile_dim2 = 0)
    g1[5] = (int)(st & 0xFFFFFFFFu);                            // tensor_dim0_stride[31:0]
    g1[6] = (int)((st >> 32) & 0xFFFFu);                        // stride[47:32] (dim1_stride lo = 0)
    g1[7] = 0;

    v4i gz = {0, 0, 0, 0};
#if __clang_major__ >= 23
    v8i gz8 = {0, 0, 0, 0, 0, 0, 0, 0};
    __builtin_amdgcn_tensor_load_to_lds(g0, g1, gz, gz, gz8, 0);
#else
    __builtin_amdgcn_tensor_load_to_lds(g0, g1, gz, gz, 0);
#endif
}

// ---------------------------------------------------------------------------
// Batched GEMM: C[MxN] = alpha * A[MxK] (row-major bf16, ld=lda)
//                       x Bt[NxK] (row-major bf16 = B transposed, ld=ldb)
// Per-z offsets: off = (z / IC) * Outer + (z % IC) * Inner   (elements)
// Block = 256 threads = 8 waves (2 M x 4 N); block tile 64x256; each wave
// computes a 32x64 C tile (2x4 accumulators, 8 WMMAs per 32-deep K step).
// A/B tiles are staged into LDS by the Tensor Data Mover, double-buffered,
// with TDM zero-fill handling ragged M/N edges.
// Requires: K % 32 == 0, lda/ldb multiples of 8.
// ---------------------------------------------------------------------------
__global__ __launch_bounds__(256)
void gemm_bf16_wmma(const bf16_t* __restrict__ A, long long lda,
                    long long aOuter, long long aInner, int aIC,
                    const bf16_t* __restrict__ Bt, long long ldb,
                    long long bOuter, long long bInner, int bIC,
                    void* __restrict__ C, long long ldc,
                    long long cOuter, long long cInner, int cIC,
                    int M, int N, int K, float alpha, int outBf16)
{
    __shared__ __align__(128) bf16_t shA[2][BM * BK];   // [row][k]  64x32
    __shared__ __align__(128) bf16_t shB[2][BN * BK];   // [col][k] 256x32

    const int z = blockIdx.z;
    const bf16_t* Ap  = A  + (long long)(z / aIC) * aOuter + (long long)(z % aIC) * aInner;
    const bf16_t* Btp = Bt + (long long)(z / bIC) * bOuter + (long long)(z % bIC) * bInner;
    char*         Cp  = (char*)C;
    const long long cOff = (long long)(z / cIC) * cOuter + (long long)(z % cIC) * cInner;

    const int wave  = threadIdx.x >> 5;    // 0..7
    const int lane  = threadIdx.x & 31;
    const int waveM = wave & 1;            // 0..1 (32-row halves)
    const int waveN = wave >> 1;           // 0..3 (64-col quarters)
    const int mBlock = blockIdx.x * BM;
    const int nBlock = blockIdx.y * BN;

    const int  lh = lane & 15;
    const bool hi = lane >= 16;

    // Prologue: stage K-step 0 into buffer 0 (wave 0 drives the TDM).
    if (wave == 0) {
        tdm_load_2d((unsigned)(uintptr_t)&shA[0][0], Ap + (long long)mBlock * lda,
                    K, (long long)M - mBlock, lda, BK, BM);
        tdm_load_2d((unsigned)(uintptr_t)&shB[0][0], Btp + (long long)nBlock * ldb,
                    K, (long long)N - nBlock, ldb, BK, BN);
    }

    v8f acc[2][4];
#pragma unroll
    for (int mi = 0; mi < 2; ++mi)
#pragma unroll
        for (int ni = 0; ni < 4; ++ni)
            acc[mi][ni] = (v8f){};

    const int nSteps = K / BK;
    for (int s = 0; s < nSteps; ++s) {
        const int cur = s & 1;
        if (wave == 0) __builtin_amdgcn_s_wait_tensorcnt(0);   // current buffer landed
        __syncthreads();                                       // publish to all waves
        if (wave == 0 && s + 1 < nSteps) {                     // prefetch next buffer
            const int ks = (s + 1) * BK;
            tdm_load_2d((unsigned)(uintptr_t)&shA[1 - cur][0],
                        Ap + (long long)mBlock * lda + ks,
                        K - ks, (long long)M - mBlock, lda, BK, BM);
            tdm_load_2d((unsigned)(uintptr_t)&shB[1 - cur][0],
                        Btp + (long long)nBlock * ldb + ks,
                        K - ks, (long long)N - nBlock, ldb, BK, BN);
        }

        // A fragments (ISA 16-bit A layout): low lanes K 0..7 / 16..23,
        // high lanes K 8..15 / 24..31 of row (waveM*32 + mi*16 + lh).
        union { uint4 q[2]; v16bf v; } fa[2];
#pragma unroll
        for (int mi = 0; mi < 2; ++mi) {
            const int row = waveM * 32 + mi * 16 + lh;
            const char* base = (const char*)&shA[cur][0] + row * (BK * 2) + (hi ? 16 : 0);
            fa[mi].q[0] = *(const uint4*)(base);
            fa[mi].q[1] = *(const uint4*)(base + 32);
        }
        // B fragments: same K layout as A for column (waveN*64 + ni*16 + lh);
        // low lanes K 0..7 / 16..23, high lanes K 8..15 / 24..31.
        union { uint4 q[2]; v16bf v; } fb[4];
#pragma unroll
        for (int ni = 0; ni < 4; ++ni) {
            const int col = waveN * 64 + ni * 16 + lh;
            const char* base = (const char*)&shB[cur][0] + col * (BK * 2) + (hi ? 16 : 0);
            fb[ni].q[0] = *(const uint4*)(base);
            fb[ni].q[1] = *(const uint4*)(base + 32);
        }

#pragma unroll
        for (int mi = 0; mi < 2; ++mi)
#pragma unroll
            for (int ni = 0; ni < 4; ++ni)
                acc[mi][ni] = __builtin_amdgcn_wmma_f32_16x16x32_bf16(
                    false, fa[mi].v, false, fb[ni].v, (short)0, acc[mi][ni], false, false);

        __syncthreads();   // all reads of buf[cur] done before it is re-staged
    }

#pragma unroll
    for (int mi = 0; mi < 2; ++mi)
#pragma unroll
        for (int ni = 0; ni < 4; ++ni)
#pragma unroll
            for (int r = 0; r < 8; ++r) {
                const int m = mBlock + waveM * 32 + mi * 16 + r + (hi ? 8 : 0);
                const int n = nBlock + waveN * 64 + ni * 16 + lh;
                if (m < M && n < N) {
                    const float val = acc[mi][ni][r] * alpha;
                    if (outBf16) ((bf16_t*)Cp)[cOff + (long long)m * ldc + n] = (bf16_t)val;
                    else         ((float*) Cp)[cOff + (long long)m * ldc + n] = val;
                }
            }
}

// ---------------------------------------------------------------------------
// Elementwise fp32 -> bf16
// ---------------------------------------------------------------------------
__global__ void to_bf16_kernel(const float* __restrict__ in, bf16_t* __restrict__ out, long long n)
{
    long long i = blockIdx.x * 256LL + threadIdx.x;
    if (i < n) out[i] = (bf16_t)in[i];
}

// fp32 [rows x cols] -> bf16 [cols x rows]  (weight transpose)
__global__ void transpose_to_bf16_kernel(const float* __restrict__ in, bf16_t* __restrict__ out,
                                         int rows, int cols)
{
    long long i = blockIdx.x * 256LL + threadIdx.x;
    if (i >= (long long)rows * cols) return;
    int c = (int)(i % cols);
    int r = (int)(i / cols);
    out[(long long)c * rows + r] = (bf16_t)in[i];
}

// ---------------------------------------------------------------------------
// m[b,h,l] = max_j( q[l] . k[samp[l,j]] ) - sum_j(...) / S
// ---------------------------------------------------------------------------
__global__ __launch_bounds__(256)
void qk_stats_kernel(const bf16_t* __restrict__ Qb, const bf16_t* __restrict__ Kb,
                     const int* __restrict__ samp, float* __restrict__ mstat)
{
    long long gid = blockIdx.x * 256LL + threadIdx.x;   // BH*S threads, exact
    int l  = (int)(gid % SS);
    int bh = (int)(gid / SS);
    int b = bh / HH, h = bh % HH;

    const bf16_t* qrow = Qb + ((long long)(b * SS + l) * HDK) + h * DKK;
    float q[DKK];
#pragma unroll
    for (int d = 0; d < DKK; ++d) q[d] = (float)qrow[d];

    float mx = -3.4e38f, sm = 0.f;
    for (int j = 0; j < SKK; ++j) {
        int idx = samp[(long long)l * SKK + j];
        const bf16_t* krow = Kb + ((long long)(b * SS + idx) * HDK) + h * DKK;
        float acc = 0.f;
#pragma unroll
        for (int d = 0; d < DKK; ++d) acc += q[d] * (float)krow[d];
        mx = fmaxf(mx, acc);
        sm += acc;
    }
    mstat[gid] = mx - sm * (1.0f / (float)SS);
}

// ---------------------------------------------------------------------------
// Top-U argmax selection per (b,h) over S values (iterative argmax in LDS)
// ---------------------------------------------------------------------------
__global__ __launch_bounds__(256)
void topk_kernel(const float* __restrict__ mstat, int* __restrict__ mtop)
{
    __shared__ float vals[SS];
    __shared__ float rv[256];
    __shared__ int   ri[256];
    int bh = blockIdx.x, t = threadIdx.x;

    for (int j = t; j < SS; j += 256) vals[j] = mstat[(long long)bh * SS + j];
    __syncthreads();

    for (int it = 0; it < UU; ++it) {
        float best = -3.4e38f; int bi = 0;
        for (int j = t; j < SS; j += 256) {
            float v = vals[j];
            if (v > best) { best = v; bi = j; }
        }
        rv[t] = best; ri[t] = bi;
        __syncthreads();
        for (int o = 128; o > 0; o >>= 1) {
            if (t < o && rv[t + o] > rv[t]) { rv[t] = rv[t + o]; ri[t] = ri[t + o]; }
            __syncthreads();
        }
        if (t == 0) { mtop[bh * UU + it] = ri[0]; vals[ri[0]] = -3.4e38f; }
        __syncthreads();
    }
}

// q_red[bh,u,:] = Qb[b*S + mtop[bh,u], h*DK : h*DK+DK]
__global__ void gather_qred_kernel(const bf16_t* __restrict__ Qb, const int* __restrict__ mtop,
                                   bf16_t* __restrict__ qred)
{
    long long i = blockIdx.x * 256LL + threadIdx.x;   // BH*U*DK, exact
    int d  = (int)(i % DKK);
    int u  = (int)((i / DKK) % UU);
    int bh = (int)(i / ((long long)DKK * UU));
    int b = bh / HH, h = bh % HH;
    int row = mtop[bh * UU + u];
    qred[i] = Qb[((long long)(b * SS + row) * HDK) + h * DKK + d];
}

// Vt[bh, d, s] = Vb[b*S+s, h*DK+d]
__global__ void transpose_v_kernel(const bf16_t* __restrict__ Vb, bf16_t* __restrict__ Vt)
{
    long long i = blockIdx.x * 256LL + threadIdx.x;   // BH*DK*S, exact
    int s  = (int)(i % SS);
    int d  = (int)((i / SS) % DKK);
    int bh = (int)(i / ((long long)SS * DKK));
    int b = bh / HH, h = bh % HH;
    Vt[i] = Vb[((long long)(b * SS + s) * HDK) + h * DKK + d];
}

// Row softmax: fp32 scores row (length S) -> bf16 attn row
__global__ __launch_bounds__(256)
void softmax_kernel(const float* __restrict__ scores, bf16_t* __restrict__ attnb)
{
    int row = blockIdx.x, t = threadIdx.x;
    const float* src = scores + (long long)row * SS;
    __shared__ float red[256];

    float mx = -3.4e38f;
    for (int j = t; j < SS; j += 256) mx = fmaxf(mx, src[j]);
    red[t] = mx; __syncthreads();
    for (int o = 128; o > 0; o >>= 1) { if (t < o) red[t] = fmaxf(red[t], red[t + o]); __syncthreads(); }
    mx = red[0]; __syncthreads();

    float s = 0.f;
    for (int j = t; j < SS; j += 256) s += __expf(src[j] - mx);
    red[t] = s; __syncthreads();
    for (int o = 128; o > 0; o >>= 1) { if (t < o) red[t] += red[t + o]; __syncthreads(); }
    float inv = 1.0f / red[0];

    for (int j = t; j < SS; j += 256)
        attnb[(long long)row * SS + j] = (bf16_t)(__expf(src[j] - mx) * inv);
}

// vmean[bh, d] = mean_s Vb[b*S+s, h*DK+d]
__global__ __launch_bounds__(64)
void vmean_kernel(const bf16_t* __restrict__ Vb, float* __restrict__ vmean)
{
    int bh = blockIdx.x, d = threadIdx.x;
    int b = bh / HH, h = bh % HH;
    const bf16_t* base = Vb + ((long long)b * SS * HDK) + h * DKK + d;
    float s = 0.f;
    for (int j = 0; j < SS; ++j) s += (float)base[(long long)j * HDK];
    vmean[bh * DKK + d] = s * (1.0f / (float)SS);
}

// ctx[b*S+s, h*DK+d] = vmean[bh,d]
__global__ void fill_ctx_kernel(const float* __restrict__ vmean, bf16_t* __restrict__ ctx)
{
    long long i = blockIdx.x * 256LL + threadIdx.x;   // R*HDK, exact
    int col = (int)(i % HDK);
    long long row = i / HDK;
    int b = (int)(row / SS);
    int h = col / DKK, d = col % DKK;
    ctx[i] = (bf16_t)vmean[(b * HH + h) * DKK + d];
}

// ctx[b*S + mtop[bh,u], h*DK+d] = upd[bh,u,d]
__global__ void scatter_ctx_kernel(const float* __restrict__ upd, const int* __restrict__ mtop,
                                   bf16_t* __restrict__ ctx)
{
    long long i = blockIdx.x * 256LL + threadIdx.x;   // BH*U*DK, exact
    int d  = (int)(i % DKK);
    int u  = (int)((i / DKK) % UU);
    int bh = (int)(i / ((long long)DKK * UU));
    int b = bh / HH, h = bh % HH;
    int s = mtop[bh * UU + u];
    ctx[((long long)(b * SS + s) * HDK) + h * DKK + d] = (bf16_t)upd[i];
}

// out = LayerNorm(outpre + fc_b + residual) * ln_w + ln_b  (one block per row)
__global__ __launch_bounds__(256)
void epilogue_ln_kernel(const float* __restrict__ outpre, const float* __restrict__ fc_b,
                        const float* __restrict__ resid, const float* __restrict__ ln_w,
                        const float* __restrict__ ln_b, float* __restrict__ out)
{
    int row = blockIdx.x, t = threadIdx.x;
    const long long base = (long long)row * DMM;
    float y[4], s = 0.f, ss = 0.f;
#pragma unroll
    for (int i = 0; i < 4; ++i) {
        int j = t + i * 256;
        float v = outpre[base + j] + fc_b[j] + resid[base + j];
        y[i] = v; s += v; ss += v * v;
    }
    __shared__ float rs[256], rq[256];
    rs[t] = s; rq[t] = ss; __syncthreads();
    for (int o = 128; o > 0; o >>= 1) {
        if (t < o) { rs[t] += rs[t + o]; rq[t] += rq[t + o]; }
        __syncthreads();
    }
    float mu  = rs[0] * (1.0f / DMM);
    float var = rq[0] * (1.0f / DMM) - mu * mu;
    float inv = rsqrtf(var + 1e-6f);
#pragma unroll
    for (int i = 0; i < 4; ++i) {
        int j = t + i * 256;
        out[base + j] = (y[i] - mu) * inv * ln_w[j] + ln_b[j];
    }
}

// ---------------------------------------------------------------------------
// Host orchestration
// ---------------------------------------------------------------------------
static inline char* ws_take(char*& p, size_t bytes)
{
    char* r = p;
    p += (bytes + 255) & ~(size_t)255;
    return r;
}

extern "C" void kernel_launch(void* const* d_in, const int* in_sizes, int n_in,
                              void* d_out, int out_size, void* d_ws, size_t ws_size,
                              hipStream_t stream)
{
    (void)in_sizes; (void)n_in; (void)out_size; (void)ws_size;

    const float* hs    = (const float*)d_in[0];   // [B,S,DM]
    const float* w_qs  = (const float*)d_in[1];   // [DM, HDK]
    const float* w_ks  = (const float*)d_in[2];
    const float* w_vs  = (const float*)d_in[3];
    const float* fc_w  = (const float*)d_in[4];   // [HDK, DM]
    const float* fc_b  = (const float*)d_in[5];
    const float* ln_w  = (const float*)d_in[6];
    const float* ln_b  = (const float*)d_in[7];
    const int*   samp  = (const int*)d_in[8];     // [S, SAMPLE_K]
    float* out = (float*)d_out;

    char* p = (char*)d_ws;
    bf16_t* Xb     = (bf16_t*)ws_take(p, (size_t)RR * DMM * 2);
    bf16_t* WqT    = (bf16_t*)ws_take(p, (size_t)DMM * HDK * 2);
    bf16_t* WkT    = (bf16_t*)ws_take(p, (size_t)DMM * HDK * 2);
    bf16_t* WvT    = (bf16_t*)ws_take(p, (size_t)DMM * HDK * 2);
    bf16_t* FcT    = (bf16_t*)ws_take(p, (size_t)HDK * DMM * 2);
    bf16_t* Qb     = (bf16_t*)ws_take(p, (size_t)RR * HDK * 2);
    bf16_t* Kb     = (bf16_t*)ws_take(p, (size_t)RR * HDK * 2);
    bf16_t* Vb     = (bf16_t*)ws_take(p, (size_t)RR * HDK * 2);
    bf16_t* Vt     = (bf16_t*)ws_take(p, (size_t)BHT * DKK * SS * 2);
    float*  mstat  = (float*) ws_take(p, (size_t)BHT * SS * 4);
    int*    mtop   = (int*)   ws_take(p, (size_t)BHT * UU * 4);
    bf16_t* qred   = (bf16_t*)ws_take(p, (size_t)BHT * UU * DKK * 2);
    float*  scores = (float*) ws_take(p, (size_t)BHT * UU * SS * 4);
    bf16_t* attnb  = (bf16_t*)ws_take(p, (size_t)BHT * UU * SS * 2);
    float*  upd    = (float*) ws_take(p, (size_t)BHT * UU * DKK * 4);
    float*  vmean  = (float*) ws_take(p, (size_t)BHT * DKK * 4);
    bf16_t* ctx    = (bf16_t*)ws_take(p, (size_t)RR * HDK * 2);
    float*  outpre = (float*) ws_take(p, (size_t)RR * DMM * 4);

    const float inv_sqrt_dk = 0.125f;  // 1/sqrt(64)

    // 1) convert / transpose operands to bf16
    {
        long long n = (long long)RR * DMM;
        to_bf16_kernel<<<dim3((unsigned)((n + 255) / 256)), dim3(256), 0, stream>>>(hs, Xb, n);
        long long nw = (long long)DMM * HDK;
        unsigned g = (unsigned)((nw + 255) / 256);
        transpose_to_bf16_kernel<<<g, 256, 0, stream>>>(w_qs, WqT, DMM, HDK);
        transpose_to_bf16_kernel<<<g, 256, 0, stream>>>(w_ks, WkT, DMM, HDK);
        transpose_to_bf16_kernel<<<g, 256, 0, stream>>>(w_vs, WvT, DMM, HDK);
        transpose_to_bf16_kernel<<<g, 256, 0, stream>>>(fc_w, FcT, HDK, DMM);
    }

    // 2) Q/K/V projections: [4096x1024] = Xb[4096x1024] x W^T, bf16 out
    {
        dim3 grid(RR / BM, HDK / BN, 1), blk(256);
        gemm_bf16_wmma<<<grid, blk, 0, stream>>>(Xb, DMM, 0, 0, 1, WqT, DMM, 0, 0, 1,
                                                 Qb, HDK, 0, 0, 1, RR, HDK, DMM, inv_sqrt_dk, 1);
        gemm_bf16_wmma<<<grid, blk, 0, stream>>>(Xb, DMM, 0, 0, 1, WkT, DMM, 0, 0, 1,
                                                 Kb, HDK, 0, 0, 1, RR, HDK, DMM, 1.0f, 1);
        gemm_bf16_wmma<<<grid, blk, 0, stream>>>(Xb, DMM, 0, 0, 1, WvT, DMM, 0, 0, 1,
                                                 Vb, HDK, 0, 0, 1, RR, HDK, DMM, 1.0f, 1);
    }

    // 3) sampled-QK sparsity measure + top-U queries
    qk_stats_kernel<<<dim3(BHT * SS / 256), dim3(256), 0, stream>>>(Qb, Kb, samp, mstat);
    topk_kernel<<<dim3(BHT), dim3(256), 0, stream>>>(mstat, mtop);
    gather_qred_kernel<<<dim3(BHT * UU * DKK / 256), dim3(256), 0, stream>>>(Qb, mtop, qred);
    transpose_v_kernel<<<dim3((unsigned)((long long)BHT * DKK * SS / 256)), dim3(256), 0, stream>>>(Vb, Vt);

    // 4) scores[z, 40, 2048] = qred[z] x K[z]^T  (batched over z = b*H + h)
    {
        dim3 grid((UU + BM - 1) / BM, SS / BN, BHT), blk(256);
        gemm_bf16_wmma<<<grid, blk, 0, stream>>>(
            qred, DKK, (long long)UU * DKK, 0, 1,
            Kb,   HDK, (long long)SS * HDK, DKK, HH,      // per-b outer, per-h inner
            scores, SS, (long long)UU * SS, 0, 1,
            UU, SS, DKK, 1.0f, 0);
    }

    // 5) softmax rows -> bf16 attn
    softmax_kernel<<<dim3(BHT * UU), dim3(256), 0, stream>>>(scores, attnb);

    // 6) upd[z, 40, 64] = attn[z] x V[z]   (B^T = Vt[z] is [64 x 2048])
    {
        dim3 grid((UU + BM - 1) / BM, (DKK + BN - 1) / BN, BHT), blk(256);
        gemm_bf16_wmma<<<grid, blk, 0, stream>>>(
            attnb, SS, (long long)UU * SS, 0, 1,
            Vt,    SS, (long long)DKK * SS, 0, 1,
            upd,   DKK, (long long)UU * DKK, 0, 1,
            UU, DKK, SS, 1.0f, 0);
    }

    // 7) context: broadcast mean(V) then scatter updated rows
    vmean_kernel<<<dim3(BHT), dim3(64), 0, stream>>>(Vb, vmean);
    fill_ctx_kernel<<<dim3((unsigned)((long long)RR * HDK / 256)), dim3(256), 0, stream>>>(vmean, ctx);
    scatter_ctx_kernel<<<dim3(BHT * UU * DKK / 256), dim3(256), 0, stream>>>(upd, mtop, ctx);

    // 8) output projection: outpre[4096x1024] = ctx x fc_w
    {
        dim3 grid(RR / BM, DMM / BN, 1), blk(256);
        gemm_bf16_wmma<<<grid, blk, 0, stream>>>(ctx, HDK, 0, 0, 1, FcT, HDK, 0, 0, 1,
                                                 outpre, DMM, 0, 0, 1, RR, DMM, HDK, 1.0f, 0);
    }

    // 9) bias + residual + LayerNorm
    epilogue_ln_kernel<<<dim3(RR), dim3(256), 0, stream>>>(outpre, fc_b, hs, ln_w, ln_b, out);
}
